// GroupedMLP_944892805464
// MI455X (gfx1250) — matrix operand order repaired and checked by reading, not compiled, not run on hardware
//
#include <hip/hip_runtime.h>
#include <hip/hip_bf16.h>
#include <math.h>

// ---------------------------------------------------------------------------
// Grouped MLP (MoE experts): out = ragged_dot(gelu(ragged_dot(x, W1)), W2)
// bf16 WMMA (v_wmma_f32_16x16x32_bf16), f32 accumulate.
// - x converted to bf16 once; both GEMMs stream A tiles with
//   global_load_async_to_lds_b128 (ASYNCcnt) straight into LDS.
// - W tiles register-prefetched, converted fp32->bf16, transposed into LDS.
// - Double-buffered LDS, BK=64 (2 WMMA K-steps per barrier).
// ws: [act bf16 128MB][x bf16 32MB][tile meta]
// ---------------------------------------------------------------------------

typedef __attribute__((ext_vector_type(16))) __bf16 v16bf;
typedef __attribute__((ext_vector_type(8)))  __bf16 bf16x8;
typedef __attribute__((ext_vector_type(4)))  __bf16 bf16x4;
typedef __attribute__((ext_vector_type(8)))  float  v8f;

constexpr int E_ = 8;
constexpr int H_ = 1024;
constexpr int F_ = 4096;
constexpr int T_ = 16384;

constexpr int BM = 128;            // tile rows (tokens)
constexpr int BN = 128;            // tile cols
constexpr int BK = 64;             // K per stage = 2 WMMA K-steps
constexpr int LDAP = BK + 8;       // padded LDS stride (elems) -> 144B rows
constexpr int MAXT = T_ / BM + E_; // 136 M-tiles max

union FragU { v16bf v; bf16x8 h[2]; };
union AccU  { v8f v;  float  f[8];  };

__device__ __forceinline__ __bf16 f2bf(float f) { return (__bf16)f; }

// --------------------------- x -> bf16 -------------------------------------
__global__ __launch_bounds__(256)
void cvt_kernel(const float* __restrict__ X, __bf16* __restrict__ Y) {
    size_t i = ((size_t)blockIdx.x * 256 + threadIdx.x) * 4;
    float4 v = *(const float4*)&X[i];
    bf16x4 p;
    p[0] = f2bf(v.x); p[1] = f2bf(v.y); p[2] = f2bf(v.z); p[3] = f2bf(v.w);
    *(bf16x4*)&Y[i] = p;
}

// --------------------------- tile schedule ---------------------------------
// meta[0] = numTiles ; then per tile: {expert, rowStart, rowCount}
__global__ void tile_map_kernel(const int* __restrict__ tokens,
                                int* __restrict__ meta) {
    if (threadIdx.x == 0 && blockIdx.x == 0) {
        int start = 0, t = 0;
        for (int e = 0; e < E_; ++e) {
            int c = tokens[e];
            for (int r = 0; r < c; r += BM) {
                meta[1 + 3 * t + 0] = e;
                meta[1 + 3 * t + 1] = start + r;
                meta[1 + 3 * t + 2] = (c - r < BM) ? (c - r) : BM;
                ++t;
            }
            start += c;
        }
        meta[0] = t;
    }
}

// --------------------------- unified grouped GEMM --------------------------
// A [.,KD] bf16 (row groups by expert), W [E][KD][ND] f32.
// GELU=true : Out = bf16 gelu(A@W)   (activation buffer)
// GELU=false: Out = f32  A@W         (final output)
template<int KD, int ND, bool GELU>
__global__ __launch_bounds__(256)
void gemm_kernel(const __bf16* __restrict__ A, const float* __restrict__ W,
                 const int* __restrict__ meta, void* __restrict__ OutPtr) {
    __shared__ __bf16 ldsA[2][BM * LDAP];
    __shared__ __bf16 ldsB[2][BN * LDAP];

    const int numTiles = meta[0];
    const int tile = blockIdx.y;           // M-tile (grouped by expert)
    if (tile >= numTiles) return;
    const int expert   = meta[1 + 3 * tile + 0];
    const int rowStart = meta[1 + 3 * tile + 1];
    const int rowCnt   = meta[1 + 3 * tile + 2];
    const int nBase    = blockIdx.x * BN;  // N-block (fastest -> W reuse in L2)

    const float* We = W + (size_t)expert * KD * ND;

    const int tid  = threadIdx.x;
    const int lane = tid & 31;
    const int wid  = tid >> 5;
    const int waveM = (wid & 3) * 32;   // 4 wave-rows * 32
    const int waveN = (wid >> 2) * 64;  // 2 wave-cols * 64
    const int fr  = lane & 15;
    const int fhi = lane >> 4;

    // B staging: thread owns column pair n2,n2+1 and k-quad kqBase(+4*it)
    const int n2     = (tid & 63) * 2;
    const int kqBase = tid >> 6;        // 0..3

    AccU acc[2][4];
    #pragma unroll
    for (int i = 0; i < 2; ++i)
        #pragma unroll
        for (int j = 0; j < 4; ++j)
            #pragma unroll
            for (int q = 0; q < 8; ++q) acc[i][j].f[q] = 0.0f;

    float2 rb[16];   // B prefetch regs: 64x128 f32 / 256 thr = 16 float2

    const unsigned long long aBase = (unsigned long long)(uintptr_t)A;

    // A tile: 128x64 bf16 = 16KB -> 4 async b128 per thread, direct to LDS.
    auto asyncLoadA = [&](int buf, int k0) {
        #pragma unroll
        for (int it = 0; it < 4; ++it) {
            int chunk = tid + it * 256;          // 0..1023
            int r  = chunk >> 3;                 // row 0..127
            int kc = (chunk & 7) * 8;            // K 0..56
            int rr = (r < rowCnt) ? r : 0;       // clamp: garbage rows unused
            unsigned ldsOff = (unsigned)(uintptr_t)&ldsA[buf][r * LDAP + kc];
            unsigned gOff = (unsigned)(((size_t)(rowStart + rr) * KD + k0 + kc)
                                       * sizeof(__bf16));
            asm volatile("global_load_async_to_lds_b128 %0, %1, %2"
                         :: "v"(ldsOff), "v"(gOff), "s"(aBase) : "memory");
        }
    };
    auto loadB = [&](int k0) {
        #pragma unroll
        for (int it = 0; it < 4; ++it) {
            int kq = kqBase + it * 4;            // 0..15 -> k quad base
            #pragma unroll
            for (int i = 0; i < 4; ++i)
                rb[it * 4 + i] = *(const float2*)
                    &We[(size_t)(k0 + 4 * kq + i) * ND + nBase + n2];
        }
    };
    auto storeB = [&](int buf) {
        #pragma unroll
        for (int it = 0; it < 4; ++it) {
            int kq = kqBase + it * 4;
            bf16x4 p0, p1;
            p0[0] = f2bf(rb[it*4+0].x); p0[1] = f2bf(rb[it*4+1].x);
            p0[2] = f2bf(rb[it*4+2].x); p0[3] = f2bf(rb[it*4+3].x);
            p1[0] = f2bf(rb[it*4+0].y); p1[1] = f2bf(rb[it*4+1].y);
            p1[2] = f2bf(rb[it*4+2].y); p1[3] = f2bf(rb[it*4+3].y);
            *(bf16x4*)&ldsB[buf][(n2 + 0) * LDAP + 4 * kq] = p0;
            *(bf16x4*)&ldsB[buf][(n2 + 1) * LDAP + 4 * kq] = p1;
        }
    };
    auto compute = [&](int buf) {
        #pragma unroll
        for (int kk = 0; kk < BK; kk += 32) {
            FragU a[2], b[4];
            #pragma unroll
            for (int i = 0; i < 2; ++i) {
                const __bf16* p =
                    &ldsA[buf][(waveM + i * 16 + fr) * LDAP + kk + 8 * fhi];
                a[i].h[0] = *(const bf16x8*)(p);
                a[i].h[1] = *(const bf16x8*)(p + 16);
            }
            #pragma unroll
            for (int j = 0; j < 4; ++j) {
                const __bf16* p =
                    &ldsB[buf][(waveN + j * 16 + fr) * LDAP + kk + 8 * fhi];
                b[j].h[0] = *(const bf16x8*)(p);
                b[j].h[1] = *(const bf16x8*)(p + 16);
            }
            #pragma unroll
            for (int i = 0; i < 2; ++i)
                #pragma unroll
                for (int j = 0; j < 4; ++j)
                    acc[i][j].v = __builtin_amdgcn_wmma_f32_16x16x32_bf16(
                        false, a[i].v, false, b[j].v, (short)0, acc[i][j].v,
                        false, false);
        }
    };

    // pipeline prologue: stage tile 0
    asyncLoadA(0, 0);
    loadB(0);
    storeB(0);
    asm volatile("s_wait_asynccnt 0x0" ::: "memory");

    int buf = 0;
    for (int k0 = 0; k0 < KD; k0 += BK) {
        __syncthreads();
        const bool hasNext = (k0 + BK) < KD;
        if (hasNext) {
            asyncLoadA(buf ^ 1, k0 + BK);  // async: HBM -> LDS, no VGPRs
            loadB(k0 + BK);                // regs (needs f32->bf16 convert)
        }
        compute(buf);
        if (hasNext) storeB(buf ^ 1);
        asm volatile("s_wait_asynccnt 0x0" ::: "memory");
        buf ^= 1;
    }

    // epilogue
    #pragma unroll
    for (int i = 0; i < 2; ++i) {
        #pragma unroll
        for (int j = 0; j < 4; ++j) {
            #pragma unroll
            for (int q = 0; q < 8; ++q) {
                int lr = waveM + i * 16 + q + 8 * fhi;
                if (lr < rowCnt) {
                    float xv = acc[i][j].f[q];
                    size_t idx = (size_t)(rowStart + lr) * ND +
                                 nBase + waveN + j * 16 + fr;
                    if constexpr (GELU) {
                        float g = 0.5f * xv *
                                  (1.0f + erff(xv * 0.70710678118654752f));
                        ((__bf16*)OutPtr)[idx] = f2bf(g);
                    } else {
                        ((float*)OutPtr)[idx] = xv;
                    }
                }
            }
        }
    }
}

// ---------------------------------------------------------------------------
extern "C" void kernel_launch(void* const* d_in, const int* in_sizes, int n_in,
                              void* d_out, int out_size, void* d_ws,
                              size_t ws_size, hipStream_t stream) {
    const float* x      = (const float*)d_in[0];
    const float* w1     = (const float*)d_in[1];
    const float* w2     = (const float*)d_in[2];
    const int*   tokens = (const int*)d_in[3];
    float* out = (float*)d_out;

    // ws layout
    constexpr size_t ACT_BYTES = (size_t)T_ * F_ * sizeof(__bf16); // 128 MB
    constexpr size_t XBF_BYTES = (size_t)T_ * H_ * sizeof(__bf16); //  32 MB
    __bf16* act = (__bf16*)d_ws;
    __bf16* xbf = (__bf16*)((char*)d_ws + ACT_BYTES);
    int*   meta = (int*)((char*)d_ws + ACT_BYTES + XBF_BYTES);

    cvt_kernel<<<dim3((T_ * H_) / (256 * 4)), dim3(256), 0, stream>>>(x, xbf);
    tile_map_kernel<<<dim3(1), dim3(32), 0, stream>>>(tokens, meta);
    gemm_kernel<H_, F_, true>
        <<<dim3(F_ / BN, MAXT), dim3(256), 0, stream>>>(xbf, w1, meta, act);
    gemm_kernel<F_, H_, false>
        <<<dim3(H_ / BN, MAXT), dim3(256), 0, stream>>>(act, w2, meta, out);
}
